// ViT_13013750907274
// MI455X (gfx1250) — matrix operand order repaired and checked by reading, not compiled, not run on hardware
//
#include <hip/hip_runtime.h>
#include <hip/hip_bf16.h>
#include <math.h>

typedef __bf16 bf16_t;
typedef __attribute__((ext_vector_type(16))) __bf16 v16bf;
typedef __attribute__((ext_vector_type(8)))  __bf16 v8bf;
typedef __attribute__((ext_vector_type(8)))  float  v8f;

#define B_   32
#define IMG_ 384
#define PH_  16
#define C_   3
#define D_   768
#define DA_  64
#define DF_  3072
#define H_   12
#define L_   12
#define NC_  1000
#define NP_  576
#define N_   577
#define G_   24              /* IMG/PH */
#define MTOK (B_ * N_)       /* 18464 */
#define MPAT (B_ * NP_)      /* 18432 */

#define CAT16(lo, hi) __builtin_shufflevector(lo, hi, 0,1,2,3,4,5,6,7,8,9,10,11,12,13,14,15)

/* ------------------------------------------------------------------ */
/* fp32 -> bf16 conversion                                             */
/* ------------------------------------------------------------------ */
__global__ void cvt_f32_bf16(const float* __restrict__ in, bf16_t* __restrict__ out, int n) {
    int i = blockIdx.x * blockDim.x + threadIdx.x;
    if (i < n) out[i] = (bf16_t)in[i];
}

/* ------------------------------------------------------------------ */
/* im2col: (B,384,384,3) -> (B*576, 768) bf16 patches                  */
/* ------------------------------------------------------------------ */
__global__ void im2col_kernel(const float* __restrict__ x, bf16_t* __restrict__ patches) {
    int i = blockIdx.x * blockDim.x + threadIdx.x;
    const int total = MPAT * D_;
    if (i >= total) return;
    int j   = i % D_;
    int row = i / D_;
    int p   = row % NP_;
    int b   = row / NP_;
    int c   = j % C_;
    int px  = (j / C_) % PH_;
    int py  = (j / C_) / PH_;
    int gx  = p % G_;
    int gy  = p / G_;
    int yy  = gy * PH_ + py;
    int xx  = gx * PH_ + px;
    patches[i] = (bf16_t)x[((long)(b * IMG_ + yy) * IMG_ + xx) * C_ + c];
}

/* ------------------------------------------------------------------ */
/* assemble h0 = concat(cls, patch_out) + pos_emb -> f32 + bf16        */
/* ------------------------------------------------------------------ */
__global__ void assemble_h0(const float* __restrict__ patch_out, const float* __restrict__ cls_token,
                            const float* __restrict__ pos_emb, float* __restrict__ h,
                            bf16_t* __restrict__ hb) {
    long i = (long)blockIdx.x * blockDim.x + threadIdx.x;
    const long total = (long)MTOK * D_;
    if (i >= total) return;
    int  d   = (int)(i % D_);
    long row = i / D_;
    int  n   = (int)(row % N_);
    long b   = row / N_;
    float v = (n == 0) ? cls_token[d] : patch_out[((long)b * NP_ + (n - 1)) * D_ + d];
    v += pos_emb[n * D_ + d];
    h[i]  = v;
    hb[i] = (bf16_t)v;
}

/* ------------------------------------------------------------------ */
/* Tiled bf16 WMMA GEMM: C(MxN) = A(MxK) @ B(KxN) + bias               */
/* Block tile 128x128, 8 waves (4x2); each wave owns 32x64 = 2x4 WMMA  */
/* tiles -> 8 v_wmma per K-step per wave.                              */
/* As: row-major [row][k] ; Bs: TRANSPOSED [col][k] so both fragment   */
/* reads are contiguous ds_load_b128 pairs.                            */
/* Requirements (hold for every call here): K % 32 == 0, Nn % 8 == 0.  */
/* mode: 0 = f32 out, 1 = bf16 out, 2 = gelu -> bf16 out               */
/* ------------------------------------------------------------------ */
#define TM 128
#define TN 128
#define TK 32

__global__ __launch_bounds__(256)
void gemm_bf16_wmma(const bf16_t* __restrict__ A, const bf16_t* __restrict__ Bm,
                    const float* __restrict__ bias, void* __restrict__ Out,
                    int M, int Nn, int K, int mode)
{
    __shared__ bf16_t As[TM * TK];     /* [row][k]              8 KB */
    __shared__ bf16_t Bs[TN * TK];     /* [col][k] (transposed) 8 KB */
    const int tid  = threadIdx.x;
    const int lane = tid & 31;
    const int wave = tid >> 5;       /* 0..7 */
    const int wm   = wave >> 1;      /* 0..3 : 32-row group */
    const int wn   = wave & 1;       /* 0..1 : 64-col group */
    const long bm  = (long)blockIdx.y * TM;
    const long bn  = (long)blockIdx.x * TN;
    const int mlo  = lane & 15;
    const int hgrp = lane >> 4;

    const v8f  vzero = {};
    const v8bf bzero = {};
    v8f acc[2][4];
    #pragma unroll
    for (int mt = 0; mt < 2; ++mt)
        #pragma unroll
        for (int nt = 0; nt < 4; ++nt) acc[mt][nt] = vzero;

    /* per-thread chunk coordinates (8 bf16 = 16 B per chunk) */
    const int ar0 = tid >> 2;              /* A rows for chunk j=0 / j=1: ar0, ar0+64 */
    const int ac  = (tid & 3) * 8;
    const int br0 = (tid >> 4);            /* B k-rows: br0, br0+16 */
    const int bc  = (tid & 15) * 8;

    for (int k0 = 0; k0 < K; k0 += TK) {
        /* ---- A tile: 512 chunks, 2 per thread, row-guarded ---- */
        #pragma unroll
        for (int j = 0; j < 2; ++j) {
            int  r  = ar0 + j * 64;
            long gr = bm + r;
            v8bf av = (gr < M) ? *(const v8bf*)(A + gr * (long)K + k0 + ac) : bzero;
            *(v8bf*)(&As[r * TK + ac]) = av;
        }
        /* ---- B tile: 512 chunks, 2 per thread, col-chunk-guarded; store transposed ---- */
        #pragma unroll
        for (int j = 0; j < 2; ++j) {
            int  r  = br0 + j * 16;        /* k row, always < K */
            long gc = bn + bc;             /* chunk start col (chunks never straddle Nn: Nn%8==0) */
            v8bf bv = (gc < Nn) ? *(const v8bf*)(Bm + (long)(k0 + r) * Nn + gc) : bzero;
            #pragma unroll
            for (int jj = 0; jj < 8; ++jj) Bs[(bc + jj) * TK + r] = bv[jj];
        }
        if (k0 + TK < K) {
            __builtin_prefetch(A + (bm + ar0) * (long)K + k0 + TK, 0, 0);
            __builtin_prefetch(Bm + (long)(k0 + TK + br0) * Nn + bn + bc, 0, 0);
        }
        __syncthreads();

        v16bf af[2], bfr[4];
        #pragma unroll
        for (int mt = 0; mt < 2; ++mt) {
            int row = wm * 32 + mt * 16 + mlo;
            v8bf lo = *(const v8bf*)(&As[row * TK + hgrp * 8]);
            v8bf hi = *(const v8bf*)(&As[row * TK + 16 + hgrp * 8]);
            af[mt] = CAT16(lo, hi);
        }
        #pragma unroll
        for (int nt = 0; nt < 4; ++nt) {
            int col = wn * 64 + nt * 16 + mlo;
            v8bf lo = *(const v8bf*)(&Bs[col * TK + hgrp * 8]);
            v8bf hi = *(const v8bf*)(&Bs[col * TK + 16 + hgrp * 8]);
            bfr[nt] = CAT16(lo, hi);
        }
        #pragma unroll
        for (int mt = 0; mt < 2; ++mt)
            #pragma unroll
            for (int nt = 0; nt < 4; ++nt)
                acc[mt][nt] = __builtin_amdgcn_wmma_f32_16x16x32_bf16(
                    false, af[mt], false, bfr[nt], (short)0, acc[mt][nt], false, false);
        __syncthreads();
    }

    /* epilogue: C/D layout row = r + 8*hgrp, col = mlo */
    #pragma unroll
    for (int mt = 0; mt < 2; ++mt)
        #pragma unroll
        for (int nt = 0; nt < 4; ++nt)
            #pragma unroll
            for (int r = 0; r < 8; ++r) {
                long gm = bm + wm * 32 + mt * 16 + r + 8 * hgrp;
                long gn = bn + wn * 64 + nt * 16 + mlo;
                if (gm < M && gn < Nn) {
                    float v = acc[mt][nt][r];
                    if (bias) v += bias[gn];
                    if (mode == 0) {
                        ((float*)Out)[gm * Nn + gn] = v;
                    } else if (mode == 1) {
                        ((bf16_t*)Out)[gm * Nn + gn] = (bf16_t)v;
                    } else {
                        v = 0.5f * v * (1.0f + erff(v * 0.70710678118654752f));
                        ((bf16_t*)Out)[gm * Nn + gn] = (bf16_t)v;
                    }
                }
            }
}

/* ------------------------------------------------------------------ */
/* Fused flash attention. qkv: (B*N, 2304) bf16 (q|k|v per head).      */
/* One workgroup = 4 waves; wave handles one 16-query tile of (b,h).   */
/* K/V streamed in blocks of 32 keys; online softmax; O via WMMA.      */
/* Ks: [key][d] (QK^T frag contiguous along d)                         */
/* Vs: [d][key] transposed (PV frag contiguous along key)              */
/* ------------------------------------------------------------------ */
#define KB 32

__global__ __launch_bounds__(128)
void attention_kernel(const bf16_t* __restrict__ qkv, bf16_t* __restrict__ o)
{
    __shared__ bf16_t Ks[KB * DA_];        /* [key][d] */
    __shared__ bf16_t Vs[DA_ * KB];        /* [d][key] */
    __shared__ bf16_t Ps[4][16 * KB];      /* per-wave P scratch, [m][k] */

    const int tid  = threadIdx.x;
    const int lane = tid & 31;
    const int wave = tid >> 5;
    const int mlo  = lane & 15;
    const int hgrp = lane >> 4;

    const int b  = blockIdx.z;
    const int h  = blockIdx.y;
    const int qt = blockIdx.x * 4 + wave;
    const bool active = (qt * 16) < N_;
    const long rowbase = (long)b * N_;
    const int qcol = h * DA_;
    const int kcol = D_ + h * DA_;
    const int vcol = 2 * D_ + h * DA_;

    const v8f vzero = {};

    /* Q fragments: clamped row index; padded rows give garbage that is
       never stored (masked at the end), so no zero-fill branch needed. */
    v16bf aq[2];
    {
        int m  = qt * 16 + mlo;
        int mc = (m < N_) ? m : (N_ - 1);
        const bf16_t* qrow = qkv + (rowbase + mc) * (long)(3 * D_) + qcol;
        #pragma unroll
        for (int ks = 0; ks < 2; ++ks) {
            v8bf lo = *(const v8bf*)(qrow + ks * 32 + hgrp * 8);
            v8bf hi = *(const v8bf*)(qrow + ks * 32 + 16 + hgrp * 8);
            aq[ks] = CAT16(lo, hi);
        }
    }

    float mrow[8], lrow[8];
    v8f accO[4];
    #pragma unroll
    for (int r = 0; r < 8; ++r) { mrow[r] = -1e30f; lrow[r] = 0.0f; }
    #pragma unroll
    for (int t = 0; t < 4; ++t) accO[t] = vzero;

    const int nKB = (N_ + KB - 1) / KB;   /* 19 */
    for (int kb = 0; kb < nKB; ++kb) {
        /* cooperative K/V block load: 256 chunks of 8, 2 per thread.
           Key index clamped: invalid keys are masked to -1e30 later,
           and their P weights are exp(-huge)=0 for the PV product.  */
        #pragma unroll
        for (int j = 0; j < 2; ++j) {
            int c  = tid + j * 128;
            int r  = c >> 3;            /* key-local 0..31 */
            int dc = (c & 7) * 8;       /* d chunk         */
            int key  = kb * KB + r;
            int keyc = (key < N_) ? key : (N_ - 1);
            long base = (rowbase + keyc) * (long)(3 * D_);
            v8bf kv = *(const v8bf*)(qkv + base + kcol + dc);
            v8bf vv = *(const v8bf*)(qkv + base + vcol + dc);
            *(v8bf*)(&Ks[r * DA_ + dc]) = kv;
            #pragma unroll
            for (int jj = 0; jj < 8; ++jj) Vs[(dc + jj) * KB + r] = vv[jj];
        }
        __syncthreads();

        if (active) {
            /* S = Q @ K^T : two 16x16 f32 tiles over the 32 keys */
            v8f sacc[2];
            sacc[0] = vzero; sacc[1] = vzero;
            #pragma unroll
            for (int nt = 0; nt < 2; ++nt)
                #pragma unroll
                for (int ks = 0; ks < 2; ++ks) {
                    const bf16_t* kp = &Ks[(nt * 16 + mlo) * DA_ + ks * 32 + hgrp * 8];
                    v8bf lo = *(const v8bf*)(kp);
                    v8bf hi = *(const v8bf*)(kp + 16);
                    v16bf bk = CAT16(lo, hi);
                    sacc[nt] = __builtin_amdgcn_wmma_f32_16x16x32_bf16(
                        false, aq[ks], false, bk, (short)0, sacc[nt], false, false);
                }

            /* scale + key-padding mask */
            float pv[2][8];
            #pragma unroll
            for (int nt = 0; nt < 2; ++nt) {
                bool kvalid = (kb * KB + nt * 16 + mlo) < N_;
                #pragma unroll
                for (int r = 0; r < 8; ++r) {
                    float s = sacc[nt][r] * 0.125f;   /* DA^-0.5 */
                    pv[nt][r] = kvalid ? s : -1e30f;
                }
            }

            /* online softmax; row m = r + 8*hgrp lives in a 16-lane half */
            #pragma unroll
            for (int r = 0; r < 8; ++r) {
                float mx = fmaxf(pv[0][r], pv[1][r]);
                #pragma unroll
                for (int off = 1; off < 16; off <<= 1)
                    mx = fmaxf(mx, __shfl_xor(mx, off, 32));
                float mnew  = fmaxf(mrow[r], mx);
                float alpha = __expf(mrow[r] - mnew);
                mrow[r] = mnew;
                float p0 = __expf(pv[0][r] - mnew);
                float p1 = __expf(pv[1][r] - mnew);
                pv[0][r] = p0; pv[1][r] = p1;
                float rs = p0 + p1;
                #pragma unroll
                for (int off = 1; off < 16; off <<= 1)
                    rs += __shfl_xor(rs, off, 32);
                lrow[r] = lrow[r] * alpha + rs;
                #pragma unroll
                for (int t = 0; t < 4; ++t)
                    accO[t][r] *= alpha;
            }

            /* D-layout -> A-layout via per-wave LDS scratch */
            #pragma unroll
            for (int nt = 0; nt < 2; ++nt)
                #pragma unroll
                for (int r = 0; r < 8; ++r)
                    Ps[wave][(r + 8 * hgrp) * KB + nt * 16 + mlo] = (bf16_t)pv[nt][r];

            v16bf ap;
            {
                v8bf lo = *(const v8bf*)(&Ps[wave][mlo * KB + hgrp * 8]);
                v8bf hi = *(const v8bf*)(&Ps[wave][mlo * KB + 16 + hgrp * 8]);
                ap = CAT16(lo, hi);
            }

            /* O += P(16x32) @ V(32x64) : 4 n-tiles */
            #pragma unroll
            for (int t = 0; t < 4; ++t) {
                const bf16_t* vp = &Vs[(t * 16 + mlo) * KB + hgrp * 8];
                v8bf lo = *(const v8bf*)(vp);
                v8bf hi = *(const v8bf*)(vp + 16);
                v16bf bv = CAT16(lo, hi);
                accO[t] = __builtin_amdgcn_wmma_f32_16x16x32_bf16(
                    false, ap, false, bv, (short)0, accO[t], false, false);
            }
        }
        __syncthreads();
    }

    if (active) {
        #pragma unroll
        for (int t = 0; t < 4; ++t)
            #pragma unroll
            for (int r = 0; r < 8; ++r) {
                int m = qt * 16 + r + 8 * hgrp;
                if (m < N_) {
                    float v = accO[t][r] / lrow[r];
                    o[(rowbase + m) * (long)D_ + h * DA_ + t * 16 + mlo] = (bf16_t)v;
                }
            }
    }
}

/* ------------------------------------------------------------------ */
/* LayerNorm (one block per row of 768).                               */
/* mode 0: custom_ln (ddof=1, 1/(std+1e-6));  mode 1: std_ln.          */
/* ------------------------------------------------------------------ */
__global__ __launch_bounds__(256)
void ln_kernel(const float* __restrict__ xin, const float* __restrict__ res,
               const float* __restrict__ g, const float* __restrict__ beta,
               float* __restrict__ fout, bf16_t* __restrict__ bout,
               long rowStride, int mode)
{
    __shared__ float red[256];
    const int  tid = threadIdx.x;
    const long row = blockIdx.x;
    const float* xr = xin + row * rowStride;
    const float* rr = res ? res + row * rowStride : nullptr;

    float tv[3];
    float s = 0.0f;
    #pragma unroll
    for (int k = 0; k < 3; ++k) {
        int i = tid + k * 256;
        float v = xr[i];
        if (rr) v += rr[i];
        tv[k] = v;
        s += v;
    }
    red[tid] = s; __syncthreads();
    for (int st = 128; st > 0; st >>= 1) { if (tid < st) red[tid] += red[tid + st]; __syncthreads(); }
    float mean = red[0] / (float)D_;
    __syncthreads();

    float s2 = 0.0f;
    #pragma unroll
    for (int k = 0; k < 3; ++k) { float d = tv[k] - mean; s2 += d * d; }
    red[tid] = s2; __syncthreads();
    for (int st = 128; st > 0; st >>= 1) { if (tid < st) red[tid] += red[tid + st]; __syncthreads(); }
    float ss = red[0];

    float scale;
    if (mode == 0) {
        float var = ss / (float)(D_ - 1);
        scale = 1.0f / (sqrtf(var) + 1e-6f);
    } else {
        float var = ss / (float)D_;
        scale = rsqrtf(var + 1e-5f);
    }

    #pragma unroll
    for (int k = 0; k < 3; ++k) {
        int i = tid + k * 256;
        float y = g[i] * (tv[k] - mean) * scale + beta[i];
        if (fout) fout[row * (long)D_ + i] = y;
        if (bout) bout[row * (long)D_ + i] = (bf16_t)y;
    }
}

/* ------------------------------------------------------------------ */
/* Host orchestration                                                  */
/* ------------------------------------------------------------------ */
extern "C" void kernel_launch(void* const* d_in, const int* in_sizes, int n_in,
                              void* d_out, int out_size, void* d_ws, size_t ws_size,
                              hipStream_t stream)
{
    (void)in_sizes; (void)n_in; (void)out_size; (void)ws_size;

    const float* x         = (const float*)d_in[0];
    const float* patch_w   = (const float*)d_in[1];
    const float* patch_b   = (const float*)d_in[2];
    const float* cls_token = (const float*)d_in[3];
    const float* pos_emb   = (const float*)d_in[4];
    const float* qkv_w     = (const float*)d_in[5];
    const float* attn_w    = (const float*)d_in[6];
    const float* attn_b    = (const float*)d_in[7];
    const float* ln1_g     = (const float*)d_in[8];
    const float* ln1_b     = (const float*)d_in[9];
    const float* ffn_w1    = (const float*)d_in[10];
    const float* ffn_b1    = (const float*)d_in[11];
    const float* ffn_w2    = (const float*)d_in[12];
    const float* ffn_b2    = (const float*)d_in[13];
    const float* ln2_g     = (const float*)d_in[14];
    const float* ln2_b     = (const float*)d_in[15];
    const float* head_g    = (const float*)d_in[16];
    const float* head_beta = (const float*)d_in[17];
    const float* head_w    = (const float*)d_in[18];
    const float* head_bias = (const float*)d_in[19];

    char* ws = (char*)d_ws;
    size_t off = 0;
    auto alloc = [&](size_t bytes) -> void* {
        void* p = ws + off;
        off += (bytes + 255) & ~(size_t)255;
        return p;
    };

    bf16_t* patchw_bf  = (bf16_t*)alloc((size_t)D_ * D_ * 2);
    bf16_t* qkvw_bf    = (bf16_t*)alloc((size_t)L_ * D_ * 3 * D_ * 2);
    bf16_t* attnw_bf   = (bf16_t*)alloc((size_t)L_ * D_ * D_ * 2);
    bf16_t* ffnw1_bf   = (bf16_t*)alloc((size_t)L_ * D_ * DF_ * 2);
    bf16_t* ffnw2_bf   = (bf16_t*)alloc((size_t)L_ * DF_ * D_ * 2);
    bf16_t* headw_bf   = (bf16_t*)alloc((size_t)D_ * NC_ * 2);
    bf16_t* patches_bf = (bf16_t*)alloc((size_t)MPAT * D_ * 2);
    float*  gemm_f32   = (float*)alloc((size_t)MTOK * D_ * 4);
    float*  h_f32      = (float*)alloc((size_t)MTOK * D_ * 4);
    bf16_t* h_bf       = (bf16_t*)alloc((size_t)MTOK * D_ * 2);
    bf16_t* qkv_bf     = (bf16_t*)alloc((size_t)MTOK * 3 * D_ * 2);
    bf16_t* o_bf       = (bf16_t*)alloc((size_t)MTOK * D_ * 2);
    bf16_t* ffn_bf     = (bf16_t*)alloc((size_t)MTOK * DF_ * 2);
    bf16_t* cls_bf     = (bf16_t*)alloc((size_t)B_ * D_ * 2);

    auto cvt = [&](const float* src, bf16_t* dst, int n) {
        cvt_f32_bf16<<<(n + 255) / 256, 256, 0, stream>>>(src, dst, n);
    };
    cvt(patch_w, patchw_bf, D_ * D_);
    cvt(qkv_w,   qkvw_bf,   L_ * D_ * 3 * D_);
    cvt(attn_w,  attnw_bf,  L_ * D_ * D_);
    cvt(ffn_w1,  ffnw1_bf,  L_ * D_ * DF_);
    cvt(ffn_w2,  ffnw2_bf,  L_ * DF_ * D_);
    cvt(head_w,  headw_bf,  D_ * NC_);

    im2col_kernel<<<(MPAT * D_ + 255) / 256, 256, 0, stream>>>(x, patches_bf);

    auto gemm = [&](const bf16_t* A, const bf16_t* Bm, const float* bias, void* Out,
                    int M, int Nn, int K, int mode) {
        dim3 grid((Nn + TN - 1) / TN, (M + TM - 1) / TM);
        gemm_bf16_wmma<<<grid, 256, 0, stream>>>(A, Bm, bias, Out, M, Nn, K, mode);
    };

    /* patch embed */
    gemm(patches_bf, patchw_bf, patch_b, gemm_f32, MPAT, D_, D_, 0);
    {
        long total = (long)MTOK * D_;
        assemble_h0<<<(int)((total + 255) / 256), 256, 0, stream>>>(
            gemm_f32, cls_token, pos_emb, h_f32, h_bf);
    }

    /* transformer layers */
    for (int l = 0; l < L_; ++l) {
        gemm(h_bf, qkvw_bf + (size_t)l * D_ * 3 * D_, nullptr, qkv_bf, MTOK, 3 * D_, D_, 1);

        dim3 agrid(10, H_, B_);   /* 37 q-tiles / 4 waves -> 10 blocks */
        attention_kernel<<<agrid, 128, 0, stream>>>(qkv_bf, o_bf);

        gemm(o_bf, attnw_bf + (size_t)l * D_ * D_, attn_b + l * D_, gemm_f32, MTOK, D_, D_, 0);
        ln_kernel<<<MTOK, 256, 0, stream>>>(gemm_f32, h_f32, ln1_g + l * D_, ln1_b + l * D_,
                                            h_f32, h_bf, (long)D_, 0);

        gemm(h_bf, ffnw1_bf + (size_t)l * D_ * DF_, ffn_b1 + l * DF_, ffn_bf, MTOK, DF_, D_, 2);
        gemm(ffn_bf, ffnw2_bf + (size_t)l * DF_ * D_, ffn_b2 + l * D_, gemm_f32, MTOK, D_, DF_, 0);
        ln_kernel<<<MTOK, 256, 0, stream>>>(gemm_f32, h_f32, ln2_g + l * D_, ln2_b + l * D_,
                                            h_f32, h_bf, (long)D_, 0);
    }

    /* head: std_ln on cls tokens, then GEMM into d_out */
    ln_kernel<<<B_, 256, 0, stream>>>(h_f32, nullptr, head_g, head_beta,
                                      nullptr, cls_bf, (long)N_ * D_, 1);
    gemm(cls_bf, headw_bf, head_bias, d_out, B_, NC_, D_, 0);
}
